// DCAlign_26774826123437
// MI455X (gfx1250) — matrix operand order repaired and checked by reading, not compiled, not run on hardware
//
#include <hip/hip_runtime.h>
#include <math.h>

// ---------------------------------------------------------------------------
// DCNv3-style deformable alignment block for MI455X (gfx1250, wave32, WMMA)
// B=8, C=256, H=W=64, K=3 (9 points). All matmuls on v_wmma_f32_16x16x32_bf16,
// each wave computing a 16x(NT*16) strip to amortize A-fragment loads.
// ---------------------------------------------------------------------------

typedef __attribute__((ext_vector_type(8)))  __bf16 v8bf;
typedef __attribute__((ext_vector_type(16))) __bf16 v16bf;
typedef __attribute__((ext_vector_type(8)))  float  v8f;

#define CC   256
#define HH   64
#define WW   64
#define HWP  4096      // H*W
#define BB   8
#define MPIX 32768     // B*H*W

__device__ __forceinline__ unsigned short f2bf(float f) {
  unsigned u = __float_as_uint(f);
  u += 0x7fffu + ((u >> 16) & 1u);       // round-to-nearest-even
  return (unsigned short)(u >> 16);
}

// ---------------------------------------------------------------------------
// 1) Pack weights: in_w/out_w fp32->bf16, off_w|mask_w -> padded (256 x 32) bf16
// ---------------------------------------------------------------------------
__global__ void pack_weights_kernel(const float* __restrict__ in_w,
                                    const float* __restrict__ out_w,
                                    const float* __restrict__ off_w,
                                    const float* __restrict__ mask_w,
                                    const float* __restrict__ off_b,
                                    const float* __restrict__ mask_b,
                                    unsigned short* __restrict__ in_wb,
                                    unsigned short* __restrict__ out_wb,
                                    unsigned short* __restrict__ om_wb,
                                    float* __restrict__ om_bias) {
  int t = blockIdx.x * blockDim.x + threadIdx.x;        // 0..65535
  if (t < CC * CC) {
    in_wb[t]  = f2bf(in_w[t]);
    out_wb[t] = f2bf(out_w[t]);
  }
  if (t < CC * 32) {
    int k = t >> 5, j = t & 31;
    float v = 0.f;
    if (j < 18)      v = off_w[k * 18 + j];
    else if (j < 27) v = mask_w[k * 9 + (j - 18)];
    om_wb[t] = f2bf(v);
  }
  if (t < 32) om_bias[t] = (t < 18) ? off_b[t] : ((t < 27) ? mask_b[t - 18] : 0.f);
}

// ---------------------------------------------------------------------------
// 2) x2 NCHW (8,256,32,32) -> NHWC (8,32,32,256) fp32
// ---------------------------------------------------------------------------
__global__ void pack_x2_kernel(const float* __restrict__ x2, float* __restrict__ x2t) {
  int idx = blockIdx.x * blockDim.x + threadIdx.x;      // (b*1024+hw)*256 + c
  int c  = idx & 255;
  int hw = (idx >> 8) & 1023;
  int b  = idx >> 18;
  x2t[idx] = x2[(b * CC + c) * 1024 + hw];
}

// ---------------------------------------------------------------------------
// 3) 2x bilinear upsample (half-pixel, edge-clamped) + |x2r[b]-x2r[b+4]|,
//    duplicated into both batch halves.  guide is NHWC fp32.
// ---------------------------------------------------------------------------
__global__ void upsample_guide_kernel(const float* __restrict__ x2t,
                                      float* __restrict__ guide) {
  int p = blockIdx.x;                                   // b<4 : 0..16383
  int c = threadIdx.x;
  int b = p >> 12;
  int h = (p >> 6) & 63, w = p & 63;
  float fy = 0.5f * (float)h - 0.25f;
  float fx = 0.5f * (float)w - 0.25f;
  float y0f = floorf(fy), x0f = floorf(fx);
  float wy = fy - y0f, wx = fx - x0f;
  int y0i = (int)y0f, x0i = (int)x0f;
  int y0 = y0i < 0 ? 0 : (y0i > 31 ? 31 : y0i);
  int y1 = (y0i + 1) < 0 ? 0 : ((y0i + 1) > 31 ? 31 : y0i + 1);
  int x0 = x0i < 0 ? 0 : (x0i > 31 ? 31 : x0i);
  int x1 = (x0i + 1) < 0 ? 0 : ((x0i + 1) > 31 ? 31 : x0i + 1);
  float w00 = (1.f - wy) * (1.f - wx), w01 = (1.f - wy) * wx;
  float w10 = wy * (1.f - wx),         w11 = wy * wx;
  const float* A  = x2t + (long)b * 1024 * CC;
  const float* Bp = x2t + (long)(b + 4) * 1024 * CC;
  float va = w00 * A[((y0 << 5) + x0) * CC + c] + w01 * A[((y0 << 5) + x1) * CC + c]
           + w10 * A[((y1 << 5) + x0) * CC + c] + w11 * A[((y1 << 5) + x1) * CC + c];
  float vb = w00 * Bp[((y0 << 5) + x0) * CC + c] + w01 * Bp[((y0 << 5) + x1) * CC + c]
           + w10 * Bp[((y1 << 5) + x0) * CC + c] + w11 * Bp[((y1 << 5) + x1) * CC + c];
  float d = fabsf(va - vb);
  guide[((long)p << 8) + c]               = d;
  guide[(((long)p + 4 * HWP) << 8) + c]   = d;
}

// ---------------------------------------------------------------------------
// 4) Depthwise 3x3 (SAME, zero pad) + LayerNorm(C) + tanh-GELU; out bf16 NHWC
// ---------------------------------------------------------------------------
__global__ void dwconv_ln_gelu_kernel(const float* __restrict__ guide,
                                      const float* __restrict__ dw_w,
                                      const float* __restrict__ dw_b,
                                      const float* __restrict__ ln_g,
                                      const float* __restrict__ ln_b,
                                      unsigned short* __restrict__ feat) {
  int p = blockIdx.x;
  int c = threadIdx.x;
  int b = p >> 12, h = (p >> 6) & 63, w = p & 63;
  float s = dw_b[c];
#pragma unroll
  for (int t = 0; t < 9; t++) {
    int dy = t / 3 - 1, dx = t % 3 - 1;
    int hh = h + dy, ww = w + dx;
    if (hh >= 0 && hh < HH && ww >= 0 && ww < WW)
      s += guide[(((long)(b * HH + hh) * WW + ww) << 8) + c] * dw_w[t * CC + c];
  }
  __shared__ float r1[CC], r2[CC];
  r1[c] = s; r2[c] = s * s;
  __syncthreads();
  for (int st = 128; st > 0; st >>= 1) {
    if (c < st) { r1[c] += r1[c + st]; r2[c] += r2[c + st]; }
    __syncthreads();
  }
  float mean = r1[0] * (1.f / 256.f);
  float var  = r2[0] * (1.f / 256.f) - mean * mean;
  float xn = (s - mean) * rsqrtf(var + 1e-5f) * ln_g[c] + ln_b[c];
  float g  = 0.5f * xn * (1.f + tanhf(0.7978845608028654f * (xn + 0.044715f * xn * xn * xn)));
  feat[((long)p << 8) + c] = f2bf(g);
}

// ---------------------------------------------------------------------------
// 5) x1 NCHW fp32 -> NHWC bf16 via 32x32 LDS transpose tiles
// ---------------------------------------------------------------------------
__global__ void pack_x1_kernel(const float* __restrict__ x1,
                               unsigned short* __restrict__ A1) {
  __shared__ float tile[32][33];
  int bt  = blockIdx.x;                 // 8 * (4096/32) * (256/32) = 8192
  int b   = bt >> 10;
  int rem = bt & 1023;
  int c0  = (rem & 7) << 5;
  int hw0 = (rem >> 3) << 5;
  int tx = threadIdx.x & 31;
  int ty = threadIdx.x >> 5;            // 0..7
#pragma unroll
  for (int i = 0; i < 4; i++) {
    int cl = ty + i * 8;
    tile[cl][tx] = x1[((long)b * CC + c0 + cl) * HWP + hw0 + tx];
  }
  __syncthreads();
#pragma unroll
  for (int i = 0; i < 4; i++) {
    int hwl = ty + i * 8;
    A1[(((long)b * HWP + hw0 + hwl) << 8) + c0 + tx] = f2bf(tile[tx][hwl]);
  }
}

// ---------------------------------------------------------------------------
// WMMA fragment loaders (A: 16x32 bf16 rows-of-M; B: 32x16 bf16 rows-of-K)
// ---------------------------------------------------------------------------
__device__ __forceinline__ v16bf load_a_frag(const unsigned short* arow, int k0, int lHi) {
  v8bf lo = *(const v8bf*)(arow + k0 + lHi);
  v8bf hi = *(const v8bf*)(arow + k0 + 16 + lHi);
  return __builtin_shufflevector(lo, hi, 0, 1, 2, 3, 4, 5, 6, 7, 8, 9, 10, 11, 12, 13, 14, 15);
}
__device__ __forceinline__ v16bf load_b_frag(const unsigned short* brow, int ldb) {
  v16bf r;
#pragma unroll
  for (int s = 0; s < 16; s++) {
    int k = (s < 8) ? s : s + 8;
    r[s] = __builtin_bit_cast(__bf16, brow[(long)k * ldb]);
  }
  return r;
}

// ---------------------------------------------------------------------------
// 6/7) WMMA GEMM: C(MxN,f32) = A(MxK,bf16) * B(KxN,bf16) + bias[N]
//      Each wave computes a 16 x (NT*16) strip: one A fragment feeds NT WMMAs
//      per K-step (NT=4 for N=256, NT=2 for the padded N=32 offset/mask GEMM).
// ---------------------------------------------------------------------------
template <int NT>
__global__ void wmma_gemm_bias_kernel(const unsigned short* __restrict__ A,
                                      const unsigned short* __restrict__ Bm,
                                      const float* __restrict__ bias,
                                      float* __restrict__ Cout,
                                      int Mtiles, int Ngroups, int K, int N) {
  int wid  = threadIdx.x >> 5;
  int lane = threadIdx.x & 31;
  int flat = blockIdx.x * 4 + wid;
  if (flat >= Mtiles * Ngroups) return;
  int tm = flat / Ngroups, tg = flat % Ngroups;
  int m0 = tm << 4, n0 = tg * (NT * 16);
  int lHi = (lane >> 4) << 3;           // 0 or 8
  int l15 = lane & 15;
  v8f acc[NT];
#pragma unroll
  for (int j = 0; j < NT; j++) acc[j] = (v8f){0.f, 0.f, 0.f, 0.f, 0.f, 0.f, 0.f, 0.f};
  const unsigned short* arow = A + (long)(m0 + l15) * K;
  for (int k0 = 0; k0 < K; k0 += 32) {
    v16bf af = load_a_frag(arow, k0, lHi);
    const unsigned short* brow = Bm + (long)(k0 + lHi) * N + n0 + l15;
#pragma unroll
    for (int j = 0; j < NT; j++) {
      v16bf bfm = load_b_frag(brow + j * 16, N);
      acc[j] = __builtin_amdgcn_wmma_f32_16x16x32_bf16(false, af, false, bfm,
                                                       (short)0, acc[j], false, false);
    }
  }
#pragma unroll
  for (int j = 0; j < NT; j++) {
    int col = n0 + j * 16 + l15;
    float bv = bias[col];
#pragma unroll
    for (int r = 0; r < 8; r++)
      Cout[(long)(m0 + lHi + r) * N + col] = acc[j][r] + bv;
  }
}

// ---------------------------------------------------------------------------
// 8) Deformable sampling: 9 points, softmax mask, bilinear zero-pad gather of
//    x_proj (NHWC fp32, L2-resident).  One block per pixel, thread = channel.
// ---------------------------------------------------------------------------
__global__ void dcn_sample_kernel(const float* __restrict__ raw,      // (M,32): 18 off | 9 mask | pad
                                  const float* __restrict__ xproj,    // (M,256) NHWC fp32
                                  unsigned short* __restrict__ accb)  // (M,256) bf16
{
  int p = blockIdx.x;
  int c = threadIdx.x;
  int b = p >> 12, h = (p >> 6) & 63, w = p & 63;
  __shared__ float sraw[27];
  if (c < 27) sraw[c] = raw[(long)p * 32 + c];
  __syncthreads();

  float mx = sraw[18];
#pragma unroll
  for (int k = 1; k < 9; k++) mx = fmaxf(mx, sraw[18 + k]);
  float ex[9]; float se = 0.f;
#pragma unroll
  for (int k = 0; k < 9; k++) { ex[k] = expf(sraw[18 + k] - mx); se += ex[k]; }
  float inv = 1.f / se;

  float acc = 0.f;
  const float* base = xproj + ((long)b * HWP << 8);
#pragma unroll
  for (int k = 0; k < 9; k++) {
    float m  = ex[k] * inv;
    float py = (float)(h + k / 3 - 1) + sraw[2 * k + 1];
    float px = (float)(w + k % 3 - 1) + sraw[2 * k];
    float y0f = floorf(py), x0f = floorf(px);
    float wy = py - y0f, wx = px - x0f;
    int y0 = (int)y0f, x0 = (int)x0f;
#pragma unroll
    for (int dy = 0; dy < 2; dy++) {
#pragma unroll
      for (int dx = 0; dx < 2; dx++) {
        int yy = y0 + dy, xx = x0 + dx;
        float wgt = (dy ? wy : 1.f - wy) * (dx ? wx : 1.f - wx);
        if (yy >= 0 && yy < HH && xx >= 0 && xx < WW)   // uniform across block
          acc += m * wgt * base[(((yy << 6) + xx) << 8) + c];
      }
    }
  }
  accb[((long)p << 8) + c] = f2bf(acc);
}

// ---------------------------------------------------------------------------
// 9) Output GEMM (16x64 strip per wave) + fused BN/ReLU/residual, NCHW store.
//    C-fragment layout: each lane owns one channel column x 8 consecutive
//    pixels -> contiguous 32B stores, no LDS transpose required.
// ---------------------------------------------------------------------------
__global__ void wmma_out_gemm_kernel(const unsigned short* __restrict__ A,   // acc bf16 (M,256)
                                     const unsigned short* __restrict__ Bm,  // out_w bf16 (256,256)
                                     const float* __restrict__ out_b,
                                     const float* __restrict__ bn_g,
                                     const float* __restrict__ bn_b,
                                     const float* __restrict__ bn_mean,
                                     const float* __restrict__ bn_var,
                                     const float* __restrict__ x1,
                                     float* __restrict__ out) {
  int wid  = threadIdx.x >> 5;
  int lane = threadIdx.x & 31;
  int flat = blockIdx.x * 4 + wid;      // Mtiles * 4 groups, grid exact
  int tg = flat & 3, tm = flat >> 2;
  int m0 = tm << 4, n0 = tg << 6;       // 64-wide N strip
  int lHi = (lane >> 4) << 3;
  int l15 = lane & 15;
  v8f acc[4];
#pragma unroll
  for (int j = 0; j < 4; j++) acc[j] = (v8f){0.f, 0.f, 0.f, 0.f, 0.f, 0.f, 0.f, 0.f};
  const unsigned short* arow = A + (long)(m0 + l15) * CC;
  for (int k0 = 0; k0 < CC; k0 += 32) {
    v16bf af = load_a_frag(arow, k0, lHi);
    const unsigned short* brow = Bm + (long)(k0 + lHi) * CC + n0 + l15;
#pragma unroll
    for (int j = 0; j < 4; j++) {
      v16bf bfm = load_b_frag(brow + j * 16, CC);
      acc[j] = __builtin_amdgcn_wmma_f32_16x16x32_bf16(false, af, false, bfm,
                                                       (short)0, acc[j], false, false);
    }
  }
  int p0 = m0 + lHi;                    // 8 consecutive pixels, same batch
  int b  = p0 >> 12, hw = p0 & 4095;
#pragma unroll
  for (int j = 0; j < 4; j++) {
    int col = n0 + j * 16 + l15;
    float scale = bn_g[col] * rsqrtf(bn_var[col] + 1e-5f);
    float shift = bn_b[col] - bn_mean[col] * scale;
    float ob    = out_b[col];
    long xbase = (long)b * CC * HWP + (long)col * HWP + hw;
#pragma unroll
    for (int r = 0; r < 8; r++) {
      float v = fmaf(acc[j][r] + ob, scale, shift);
      v = fmaxf(v, 0.f);
      out[xbase + r] = v + x1[xbase + r];
    }
  }
}

// ---------------------------------------------------------------------------
// Host side
// ---------------------------------------------------------------------------
extern "C" void kernel_launch(void* const* d_in, const int* in_sizes, int n_in,
                              void* d_out, int out_size, void* d_ws, size_t ws_size,
                              hipStream_t stream) {
  const float* x1      = (const float*)d_in[0];
  const float* x2      = (const float*)d_in[1];
  const float* dw_w    = (const float*)d_in[2];
  const float* dw_b    = (const float*)d_in[3];
  const float* ln_g    = (const float*)d_in[4];
  const float* ln_b    = (const float*)d_in[5];
  const float* off_w   = (const float*)d_in[6];
  const float* off_b   = (const float*)d_in[7];
  const float* mask_w  = (const float*)d_in[8];
  const float* mask_b  = (const float*)d_in[9];
  const float* in_w    = (const float*)d_in[10];
  const float* in_b    = (const float*)d_in[11];
  const float* out_w   = (const float*)d_in[12];
  const float* out_b   = (const float*)d_in[13];
  const float* bn_g    = (const float*)d_in[14];
  const float* bn_b    = (const float*)d_in[15];
  const float* bn_mean = (const float*)d_in[16];
  const float* bn_var  = (const float*)d_in[17];
  float* out = (float*)d_out;

  // Workspace layout (256B aligned regions, aliased where lifetimes allow):
  char* ws = (char*)d_ws;
  size_t o = 0;
  auto take = [&](size_t bytes) -> char* {
    char* p = ws + o;
    o = (o + bytes + 255) & ~(size_t)255;
    return p;
  };
  float*          gx      = (float*)take((size_t)MPIX * CC * 4);  // guide, then x_proj
  unsigned short* feat    = (unsigned short*)take((size_t)MPIX * CC * 2);
  unsigned short* a1_acc  = (unsigned short*)take((size_t)MPIX * CC * 2);  // x1 bf16, then acc bf16
  float*          offmask = (float*)take((size_t)MPIX * 32 * 4);
  unsigned short* in_wb   = (unsigned short*)take((size_t)CC * CC * 2);
  unsigned short* out_wb  = (unsigned short*)take((size_t)CC * CC * 2);
  unsigned short* om_wb   = (unsigned short*)take((size_t)CC * 32 * 2);
  float*          om_bias = (float*)take(32 * 4);
  float*          x2t     = (float*)take((size_t)BB * CC * 1024 * 4);

  pack_weights_kernel<<<256, 256, 0, stream>>>(in_w, out_w, off_w, mask_w, off_b,
                                               mask_b, in_wb, out_wb, om_wb, om_bias);
  pack_x2_kernel<<<8192, 256, 0, stream>>>(x2, x2t);
  upsample_guide_kernel<<<4 * HWP, 256, 0, stream>>>(x2t, gx);
  dwconv_ln_gelu_kernel<<<MPIX, 256, 0, stream>>>(gx, dw_w, dw_b, ln_g, ln_b, feat);
  pack_x1_kernel<<<8192, 256, 0, stream>>>(x1, a1_acc);

  // x_proj = x1_nhwc @ in_w + in_b   (M=32768, N=256, K=256) -> overwrites guide
  // waves = Mtiles(2048) * Ngroups(4) = 8192 -> 2048 blocks of 4 waves
  wmma_gemm_bias_kernel<4><<<2048, 128, 0, stream>>>(
      a1_acc, in_wb, in_b, gx, 2048, 4, 256, 256);
  // offsets/mask logits = feat @ [off_w|mask_w|0] + [off_b|mask_b|0]  (N padded 32)
  // waves = 2048 * 1 -> 512 blocks
  wmma_gemm_bias_kernel<2><<<512, 128, 0, stream>>>(
      feat, om_wb, om_bias, offmask, 2048, 1, 256, 32);

  // deformable sampling -> acc bf16 (overwrites x1 bf16)
  dcn_sample_kernel<<<MPIX, 256, 0, stream>>>(offmask, gx, a1_acc);

  // y = acc @ out_w + out_b, BN + ReLU + residual, NCHW  (2048*4 waves)
  wmma_out_gemm_kernel<<<2048, 128, 0, stream>>>(
      a1_acc, out_wb, out_b, bn_g, bn_b, bn_mean, bn_var, x1, out);
  (void)in_sizes; (void)n_in; (void)out_size; (void)ws_size;
}
